// Zeroshot_66511863546066
// MI455X (gfx1250) — compile-verified, matrix operand order checked by reading, not run on hardware
//
#include <hip/hip_runtime.h>
#include <hip/hip_bf16.h>

typedef __attribute__((ext_vector_type(16))) _Float16 v16h;
typedef __attribute__((ext_vector_type(8)))  _Float16 v8h;
typedef __attribute__((ext_vector_type(8)))  float    v8f;

#define B_  4
#define S_  16
#define N_  32
#define DD_ 100

// ---------- helpers ----------
__device__ __forceinline__ v16h ldA(const _Float16* p) {
    // lane's A fragment: halfs [0..7] at p, [8..15] at p+16 (K split per ISA layout)
    v8h lo = *(const v8h*)p;
    v8h hi = *(const v8h*)(p + 16);
    v16h r;
#pragma unroll
    for (int i = 0; i < 8; ++i) { r[i] = lo[i]; r[8 + i] = hi[i]; }
    return r;
}
__device__ __forceinline__ v16h ldB(const _Float16* p) {
    return *(const v16h*)p;  // 32B contiguous, pre-swizzled
}

// Pipelined GEMM inner loop: batch all A fragments (one ds clause / wait),
// prefetch next B tile during WMMA execution, WMMAs issue back-to-back.
template <int MT, int KS, int STRIDE>
__device__ __forceinline__ void gemm_tile(const _Float16* __restrict__ src,
                                          const _Float16* __restrict__ W,
                                          int nt, int lane, int l, int lhalf,
                                          v8f* acc) {
    const _Float16* abase = src + l * STRIDE + lhalf * 8;
    v16h bm = ldB(W + ((size_t)((nt * KS + 0) * 32 + lane) << 4));
#pragma unroll 1
    for (int ks = 0; ks < KS; ++ks) {
        v16h am[MT];
#pragma unroll
        for (int mt = 0; mt < MT; ++mt)
            am[mt] = ldA(abase + mt * 16 * STRIDE + ks * 32);
        v16h bnext = bm;
        if (ks + 1 < KS)
            bnext = ldB(W + ((size_t)((nt * KS + ks + 1) * 32 + lane) << 4));
#pragma unroll
        for (int mt = 0; mt < MT; ++mt)
            acc[mt] = __builtin_amdgcn_wmma_f32_16x16x32_f16(
                false, am[mt], false, bm, (short)0, acc[mt], false, false);
        bm = bnext;
    }
}

// ---------- weight swizzle: W(K x 256) f32 -> per-lane B-operand f16 ----------
// layout: out[(((nt*KS + ks)*32 + lane)*16) + i]
__global__ __launch_bounds__(256) void kprep(const float* __restrict__ W, int Ksrc, int KS,
                                             _Float16* __restrict__ out, int total) {
    int f = blockIdx.x * 256 + threadIdx.x;
    if (f >= total) return;
    int i    = f & 15;
    int lane = (f >> 4) & 31;
    int rest = f >> 9;
    int ks   = rest % KS;
    int nt   = rest / KS;
    int h    = lane >> 4, l = lane & 15;
    int n    = nt * 16 + l;
    int kl   = (i < 8) ? (h * 8 + i) : (16 + h * 8 + (i - 8));
    int k    = ks * 32 + kl;
    float v  = (k < Ksrc) ? W[k * 256 + n] : 0.f;
    out[f]   = (_Float16)v;
}

// ---------- k1: att MLP + ha = att_f @ gW1[514:] ----------
__global__ __launch_bounds__(256) void k1(const float* __restrict__ att,
                                          const float* __restrict__ attW1, const float* __restrict__ attb1,
                                          const float* __restrict__ attW2, const float* __restrict__ attb2,
                                          const float* __restrict__ gW1,   float* __restrict__ ha) {
    __shared__ float satt[312];
    __shared__ float sa[512];
    __shared__ float sf[514];
    int bn = blockIdx.x;  // b*32+n
    const float* ar = att + bn * 312;
    int t = threadIdx.x;
    for (int e = t; e < 312; e += 256) satt[e] = ar[e];
    __syncthreads();
    for (int o = t; o < 512; o += 256) {
        float s = attb1[o];
        for (int k = 0; k < 312; ++k) s = fmaf(satt[k], attW1[k * 512 + o], s);
        sa[o] = fmaxf(s, 0.f);
    }
    __syncthreads();
    for (int o = t; o < 514; o += 256) {
        float s = attb2[o];
        for (int k = 0; k < 512; ++k) s = fmaf(sa[k], attW2[k * 514 + o], s);
        sf[o] = fmaxf(s, 0.f);
    }
    __syncthreads();
    for (int o = t; o < 256; o += 256) {
        float s = 0.f;
        for (int k = 0; k < 514; ++k) s = fmaf(sf[k], gW1[(514 + k) * 256 + o], s);
        ha[bn * 256 + o] = s;
    }
}

// ---------- k2: hx[b,s] = xc(100x514, padded 112x544) @ Wx  (WMMA) ----------
#define K2_STRIDE 552  // 544 + 8 halfs pad
__global__ __launch_bounds__(128) void k2(const float* __restrict__ x_f,
                                          const _Float16* __restrict__ swW1x,
                                          float* __restrict__ hx) {
    extern __shared__ _Float16 smem2[];  // 112 * 552 halfs
    int bs = blockIdx.x;                 // b*16+s
    const float* xr = x_f + (size_t)bs * 512 * DD_;
    int tid = threadIdx.x;
    // bulk zero (covers all pad rows/cols), wide stores
    {
        v8h z = {};
        v8h* zp = (v8h*)smem2;  // 112*552/8 = 7728 vectors
        for (int e = tid; e < 112 * K2_STRIDE / 8; e += 128) zp[e] = z;
    }
    __syncthreads();
    // coalesced transpose fill: lanes walk p (contiguous in x_f), c fixed per pass
    for (int e = tid; e < 512 * 128; e += 128) {
        int c = e >> 7, p = e & 127;
        if (p < DD_) smem2[p * K2_STRIDE + c] = (_Float16)xr[c * DD_ + p];
    }
    for (int e = tid; e < 2 * 128; e += 128) {
        int c = 512 + (e >> 7), p = e & 127;
        if (p < DD_)
            smem2[p * K2_STRIDE + c] = (_Float16)(((c == 512) ? (p / 10) : (p % 10)) * 0.1f);
    }
    __syncthreads();
    int wave = tid >> 5, lane = tid & 31;
    int lhalf = lane >> 4, l = lane & 15;
    float* outr = hx + (size_t)bs * DD_ * 256;
#pragma unroll 1
    for (int j = 0; j < 4; ++j) {
        int nt = wave * 4 + j;
        v8f acc[7];
#pragma unroll
        for (int mt = 0; mt < 7; ++mt) { v8f z = {}; acc[mt] = z; }
        gemm_tile<7, 17, K2_STRIDE>(smem2, swW1x, nt, lane, l, lhalf, acc);
#pragma unroll
        for (int mt = 0; mt < 7; ++mt)
#pragma unroll
            for (int i = 0; i < 8; ++i) {
                int row = mt * 16 + lhalf * 8 + i;
                if (row < DD_) outr[row * 256 + nt * 16 + l] = acc[mt][i];
            }
    }
}

// ---------- k3: per (b,s,n) fused 3-layer MLP + DD-sum (WMMA) ----------
#define K3_STRIDE 264  // 256 + 8 halfs pad
#define K3_BUF (112 * K3_STRIDE)
__global__ __launch_bounds__(128) void k3(const float* __restrict__ hx,
                                          const float* __restrict__ ha,
                                          const float* __restrict__ gb1,
                                          const _Float16* __restrict__ swW2,
                                          const _Float16* __restrict__ swW3,
                                          const _Float16* __restrict__ swW4,
                                          const float* __restrict__ gb2,
                                          const float* __restrict__ gb3,
                                          const float* __restrict__ gb4,
                                          float* __restrict__ hsum) {
    extern __shared__ _Float16 smem3[];  // 2 * K3_BUF halfs
    _Float16* bufA = smem3;
    _Float16* bufB = smem3 + K3_BUF;
    int wg = blockIdx.x;          // ((b*S+s)*N + n)
    int bs = wg >> 5, n = wg & 31;
    int b  = bs >> 4;
    const float* hxr = hx + (size_t)bs * DD_ * 256;
    const float* har = ha + (size_t)(b * N_ + n) * 256;
    int tid = threadIdx.x;
    // stage 0: H0 = relu(hx + ha + gb1) -> f16 LDS
    for (int e = tid; e < DD_ * 256; e += 128) {
        int p = e >> 8, c = e & 255;
        float v = hxr[e] + har[c] + gb1[c];
        bufA[p * K3_STRIDE + c] = (_Float16)fmaxf(v, 0.f);
    }
    for (int e = tid; e < 12 * 256; e += 128) {
        int p = DD_ + (e >> 8), c = e & 255;
        bufA[p * K3_STRIDE + c] = (_Float16)0.f;
    }
    __syncthreads();
    int wave = tid >> 5, lane = tid & 31;
    int lhalf = lane >> 4, l = lane & 15;
    const _Float16* swW[3] = {swW2, swW3, swW4};
    const float*    bv3[3] = {gb2, gb3, gb4};
#pragma unroll 1
    for (int layer = 0; layer < 3; ++layer) {
        const _Float16* src = (layer & 1) ? bufB : bufA;
        _Float16*       dst = (layer & 1) ? bufA : bufB;
        const _Float16* W   = swW[layer];
#pragma unroll 1
        for (int j = 0; j < 4; ++j) {
            int nt = wave * 4 + j;
            v8f acc[7];
#pragma unroll
            for (int mt = 0; mt < 7; ++mt) { v8f z = {}; acc[mt] = z; }
            gemm_tile<7, 8, K3_STRIDE>(src, W, nt, lane, l, lhalf, acc);
            float bv = bv3[layer][nt * 16 + l];
            if (layer < 2) {
#pragma unroll
                for (int mt = 0; mt < 7; ++mt)
#pragma unroll
                    for (int i = 0; i < 8; ++i) {
                        int row = mt * 16 + lhalf * 8 + i;
                        float v = fmaxf(acc[mt][i] + bv, 0.f);
                        if (row >= DD_) v = 0.f;  // keep pad rows zero
                        dst[row * K3_STRIDE + nt * 16 + l] = (_Float16)v;
                    }
            } else {
                float psum = 0.f;
#pragma unroll
                for (int mt = 0; mt < 7; ++mt)
#pragma unroll
                    for (int i = 0; i < 8; ++i) {
                        int row = mt * 16 + lhalf * 8 + i;
                        if (row < DD_) psum += fmaxf(acc[mt][i] + bv, 0.f);
                    }
                psum += __shfl_xor(psum, 16);  // fold lane+16 (same col, other row half)
                if (lane < 16) hsum[(size_t)wg * 256 + nt * 16 + l] = psum;
            }
        }
        __syncthreads();
    }
}

// ---------- k4w: WMMA head (fc1+fc2) + scalar 29-wide tail + loss ----------
#define K4_ROWS 128
__global__ __launch_bounds__(128) void k4w(const float* __restrict__ hsum,
                                           const _Float16* __restrict__ swF1, const float* __restrict__ fb1,
                                           const _Float16* __restrict__ swF2, const float* __restrict__ fb2,
                                           const float* __restrict__ fW3, const float* __restrict__ fb3,
                                           const float* __restrict__ gmm, const float* __restrict__ beta,
                                           const float* __restrict__ mean, const float* __restrict__ var,
                                           const float* __restrict__ fW4, const float* __restrict__ fb4,
                                           const int* __restrict__ x_label, const int* __restrict__ att_label,
                                           float* __restrict__ partial) {
    extern __shared__ _Float16 smem4[];
    _Float16* bufA = smem4;                       // 128 x 264 halfs
    _Float16* bufB = smem4 + K4_ROWS * K3_STRIDE;
    float*    ybuf = (float*)(smem4 + 2 * K4_ROWS * K3_STRIDE);  // 128 x 29 f32
    int wg = blockIdx.x, tid = threadIdx.x;
    const float* hr = hsum + (size_t)wg * K4_ROWS * 256;
    for (int e = tid; e < K4_ROWS * 256; e += 128) {
        int p = e >> 8, c = e & 255;
        bufA[p * K3_STRIDE + c] = (_Float16)hr[e];
    }
    __syncthreads();
    int wave = tid >> 5, lane = tid & 31;
    int lhalf = lane >> 4, l = lane & 15;
    const _Float16* swW[2] = {swF1, swF2};
    const float*    bb[2]  = {fb1, fb2};
#pragma unroll 1
    for (int layer = 0; layer < 2; ++layer) {
        const _Float16* src = layer ? bufB : bufA;
        _Float16*       dst = layer ? bufA : bufB;
#pragma unroll 1
        for (int j = 0; j < 4; ++j) {
            int nt = wave * 4 + j;
            v8f acc[8];
#pragma unroll
            for (int mt = 0; mt < 8; ++mt) { v8f z = {}; acc[mt] = z; }
            gemm_tile<8, 8, K3_STRIDE>(src, swW[layer], nt, lane, l, lhalf, acc);
            float bv = bb[layer][nt * 16 + l];
#pragma unroll
            for (int mt = 0; mt < 8; ++mt)
#pragma unroll
                for (int i = 0; i < 8; ++i) {
                    int row = mt * 16 + lhalf * 8 + i;
                    dst[row * K3_STRIDE + nt * 16 + l] = (_Float16)fmaxf(acc[mt][i] + bv, 0.f);
                }
        }
        __syncthreads();
    }
    // bufA now holds h2 (f16). fc3 (29 cols) + BN + relu
    for (int e = tid; e < K4_ROWS * 29; e += 128) {
        int row = e / 29, o = e - row * 29;
        float s = fb3[o];
        for (int k = 0; k < 256; ++k) s = fmaf((float)bufA[row * K3_STRIDE + k], fW3[k * 29 + o], s);
        s = (s - mean[o]) * rsqrtf(var[o] + 1e-5f) * gmm[o] + beta[o];
        ybuf[row * 29 + o] = fmaxf(s, 0.f);
    }
    __syncthreads();
    {
        int rg = wg * K4_ROWS + tid;  // global row, 128 threads = 128 rows
        float z = fb4[0];
        for (int k = 0; k < 29; ++k) z = fmaf(ybuf[tid * 29 + k], fW4[k], z);
        float sc = 1.f / (1.f + expf(-z));
        int b = rg >> 9, rem = rg & 511, sr = rem >> 5, n = rem & 31;
        float lab = (x_label[b * S_ + sr] == att_label[b * N_ + n]) ? 1.f : 0.f;
        float d = lab - sc;
        partial[rg] = d * d;
    }
}

// ---------- deterministic reduction ----------
__global__ __launch_bounds__(256) void kred(const float* __restrict__ partial, float* __restrict__ out) {
    __shared__ float s[256];
    int t = threadIdx.x;
    float a = 0.f;
    for (int e = t; e < 2048; e += 256) a += partial[e];
    s[t] = a;
    __syncthreads();
    for (int w = 128; w > 0; w >>= 1) {
        if (t < w) s[t] += s[t + w];
        __syncthreads();
    }
    if (t == 0) out[0] = s[0];
}

extern "C" void kernel_launch(void* const* d_in, const int* in_sizes, int n_in,
                              void* d_out, int out_size, void* d_ws, size_t ws_size,
                              hipStream_t stream) {
    (void)in_sizes; (void)n_in; (void)out_size; (void)ws_size;
    const float* x_f     = (const float*)d_in[0];
    const float* att     = (const float*)d_in[1];
    const int*   x_label = (const int*)d_in[2];
    const int*   att_lab = (const int*)d_in[3];
    const float* attW1   = (const float*)d_in[4];
    const float* attb1   = (const float*)d_in[5];
    const float* attW2   = (const float*)d_in[6];
    const float* attb2   = (const float*)d_in[7];
    const float* gW1     = (const float*)d_in[8];
    const float* gb1     = (const float*)d_in[9];
    const float* gW2     = (const float*)d_in[10];
    const float* gb2     = (const float*)d_in[11];
    const float* gW3     = (const float*)d_in[12];
    const float* gb3     = (const float*)d_in[13];
    const float* gW4     = (const float*)d_in[14];
    const float* gb4     = (const float*)d_in[15];
    const float* fW1     = (const float*)d_in[16];
    const float* fb1     = (const float*)d_in[17];
    const float* fW2     = (const float*)d_in[18];
    const float* fb2     = (const float*)d_in[19];
    const float* fW3     = (const float*)d_in[20];
    const float* fb3     = (const float*)d_in[21];
    const float* bn_g    = (const float*)d_in[22];
    const float* bn_b    = (const float*)d_in[23];
    const float* bn_m    = (const float*)d_in[24];
    const float* bn_v    = (const float*)d_in[25];
    const float* fW4     = (const float*)d_in[26];
    const float* fb4     = (const float*)d_in[27];

    // workspace layout
    float* ws_f   = (float*)d_ws;
    float* ha     = ws_f;                       // 32768 floats
    float* hx     = ws_f + 32768;               // 1,638,400 floats
    float* hsum   = ws_f + 32768 + 1638400;     // 524,288 floats
    float* part   = ws_f + 2195456;             // 2048 floats
    _Float16* wsH = (_Float16*)((char*)d_ws + (size_t)8790016);
    _Float16* swW1x = wsH;                      // 16*17*512 = 139264 halfs
    _Float16* swW2  = swW1x + 139264;           // 65536 halfs each below
    _Float16* swW3  = swW2 + 65536;
    _Float16* swW4  = swW3 + 65536;
    _Float16* swF1  = swW4 + 65536;
    _Float16* swF2  = swF1 + 65536;

    // 1) weight swizzles
    kprep<<<(139264 + 255) / 256, 256, 0, stream>>>(gW1, 514, 17, swW1x, 139264);
    kprep<<<256, 256, 0, stream>>>(gW2, 256, 8, swW2, 65536);
    kprep<<<256, 256, 0, stream>>>(gW3, 256, 8, swW3, 65536);
    kprep<<<256, 256, 0, stream>>>(gW4, 256, 8, swW4, 65536);
    kprep<<<256, 256, 0, stream>>>(fW1, 256, 8, swF1, 65536);
    kprep<<<256, 256, 0, stream>>>(fW2, 256, 8, swF2, 65536);
    // 2) attention MLP -> ha
    k1<<<B_ * N_, 256, 0, stream>>>(att, attW1, attb1, attW2, attb2, gW1, ha);
    // 3) hx GEMM (WMMA)
    k2<<<B_ * S_, 128, (size_t)112 * K2_STRIDE * sizeof(_Float16), stream>>>(x_f, swW1x, hx);
    // 4) fused 3-layer MLP + sum (WMMA)
    k3<<<B_ * S_ * N_, 128, (size_t)2 * K3_BUF * sizeof(_Float16), stream>>>(
        hx, ha, gb1, swW2, swW3, swW4, gb2, gb3, gb4, hsum);
    // 5) head: fc1+fc2 WMMA, 29-wide tail scalar, loss partials
    size_t lds4 = (size_t)2 * K4_ROWS * K3_STRIDE * sizeof(_Float16) + (size_t)K4_ROWS * 29 * sizeof(float);
    k4w<<<16, 128, lds4, stream>>>(hsum, swF1, fb1, swF2, fb2, fW3, fb3,
                                   bn_g, bn_b, bn_m, bn_v, fW4, fb4,
                                   x_label, att_lab, part);
    // 6) deterministic reduce
    kred<<<1, 256, 0, stream>>>(part, (float*)d_out);
}
